// QuanvolutionClassifier_65481071403850
// MI455X (gfx1250) — matrix operand order repaired and checked by reading, not compiled, not run on hardware
//
#include <hip/hip_runtime.h>
#include <hip/hip_bf16.h>

typedef __attribute__((ext_vector_type(16))) _Float16 v16h;
typedef __attribute__((ext_vector_type(8)))  _Float16 v8h;
typedef __attribute__((ext_vector_type(8)))  float    v8f;

#define ROWS    16
#define KREAL   784
#define NWAVES  4
#define NTHREADS (NWAVES * 32)
#define KPAD    896   // 28 * 32 ; divisible by 32*NWAVES -> branch-free wave split
#define KBLOCKS (KPAD / 32)            // 28
#define QITER   (KBLOCKS / NWAVES)     // 7 per wave, uniform
#define KSTRIDE 904   // 1808B row = 452 dwords; 452%64=4 -> conflict-free b128

__global__ __launch_bounds__(NTHREADS)
void quanv_fused_kernel(const float* __restrict__ x,
                        const float* __restrict__ conv_w,
                        const float* __restrict__ conv_b,
                        const float* __restrict__ bn_gamma,
                        const float* __restrict__ bn_beta,
                        const float* __restrict__ bn_mean,
                        const float* __restrict__ bn_var,
                        const float* __restrict__ shortcut_weight,
                        const float* __restrict__ theta,
                        const float* __restrict__ lin_w,
                        const float* __restrict__ lin_b,
                        const float* __restrict__ log_scale,
                        float* __restrict__ out,
                        int Btotal)
{
    // row-major [M][K] / [N][K] tiles: fragment halves contiguous -> ds_load_b128
    __shared__ __align__(16) _Float16 ftile[ROWS][KSTRIDE];
    __shared__ __align__(16) _Float16 btile[16][KSTRIDE];
    __shared__ float pltile[NWAVES][ROWS][16];   // per-wave partial C tiles

    const int tid  = threadIdx.x;
    const int lane = tid & 31;
    const int wave = tid >> 5;
    const int row0 = blockIdx.x * ROWS;
    (void)Btotal;

    // ---- uniform parameters (folded BN + shortcut scale) ----
    const float sw = shortcut_weight[0];
    float wc[4][4], Ac[4], Bc[4];
    #pragma unroll
    for (int c = 0; c < 4; ++c) {
        float rs = rsqrtf(bn_var[c] + 1e-5f);
        float a  = bn_gamma[c] * rs;
        Ac[c] = sw * a;
        Bc[c] = sw * ((conv_b[c] - bn_mean[c]) * a + bn_beta[c]);
        #pragma unroll
        for (int q = 0; q < 4; ++q) wc[c][q] = conv_w[c * 4 + q];
    }
    float thh[4], cpo[4], spo[4];
    #pragma unroll
    for (int w = 0; w < 4; ++w) {
        thh[w] = theta[w] * 0.5f;
        float g = theta[4 + w] * 0.5f;
        cpo[w] = __cosf(g);
        spo[w] = __sinf(g);
    }

    // ---- stage lin_w row-major + zero-padded into LDS as f16 [16][KSTRIDE] ----
    for (int t = tid; t < 16 * KPAD; t += NTHREADS) {           // 112 trips, uniform
        int n = t / KPAD, k = t - n * KPAD;
        float v = (k < KREAL && n < 10) ? lin_w[n * KREAL + k] : 0.0f;
        btile[n][k] = (_Float16)v;
    }
    // ---- zero K-padding of feature tile: k in [784,896), 16 rows ----
    for (int t = tid; t < ROWS * (KPAD - KREAL); t += NTHREADS) { // 14 trips
        int r = t / (KPAD - KREAL);
        int k = KREAL + (t - r * (KPAD - KREAL));
        ftile[r][k] = (_Float16)0.0f;
    }

    // ---- Phase A: quanvolution, one 2-qubit pair per item (no redundant trig) --
    // domain: r(16) x p(196) x h(2) = 6272 = 128*49, uniform trips
    for (int t = tid; t < ROWS * 196 * 2; t += NTHREADS) {
        int r   = t / 392;
        int rem = t - r * 392;
        int p   = rem >> 1;
        int h   = rem & 1;                 // 0: wires(0,1), 1: wires(2,3)
        int i2  = p / 14;
        int j2  = p - i2 * 14;
        int w0  = h << 1;
        const float* qp = x + (size_t)(row0 + r) * KREAL + (2 * i2 + h) * 28 + 2 * j2;
        float aa = qp[0] * 0.5f + thh[w0];       // RY composes angles
        float bb = qp[1] * 0.5f + thh[w0 + 1];
        float ca = __cosf(aa), sa = __sinf(aa);
        float cb = __cosf(bb), sb = __sinf(bb);
        // state after CNOT(control=a, target=b)
        float w00 = ca * cb, w01 = ca * sb, w10 = sa * sb, w11 = sa * cb;
        float c1 = cpo[w0], s1 = spo[w0], c2 = cpo[w0 + 1], s2 = spo[w0 + 1];
        float u00 = c1 * w00 - s1 * w10;
        float u01 = c1 * w01 - s1 * w11;
        float u10 = s1 * w00 + c1 * w10;
        float u11 = s1 * w01 + c1 * w11;
        float v00 = c2 * u00 - s2 * u01;
        float v01 = s2 * u00 + c2 * u01;
        float v10 = c2 * u10 - s2 * u11;
        float v11 = s2 * u10 + c2 * u11;
        float p00 = v00 * v00, p01 = v01 * v01, p10 = v10 * v10, p11 = v11 * v11;
        float Za = p00 + p01 - p10 - p11;
        float Zb = p00 - p01 + p10 - p11;
        int k = (p << 2) + (h << 1);
        ftile[r][k]     = (_Float16)Za;    // meas m = 2h
        ftile[r][k + 1] = (_Float16)Zb;    // meas m = 2h+1
    }

    __syncthreads();

    // ---- Phase B: shortcut conv+BN, one site -> 4 channels (shared pixels) ----
    // domain: r(16) x site(196) = 3136; accumulate into ftile (per-thread RMW)
    for (int t = tid; t < ROWS * 196; t += NTHREADS) {
        int r  = t / 196;
        int s  = t - r * 196;              // s = i*14 + j
        int i  = s / 14;
        int j  = s - i * 14;
        const float* xp = x + (size_t)(row0 + r) * KREAL + (2 * i) * 28 + 2 * j;
        float p0 = xp[0], p1 = xp[1], p2 = xp[28], p3 = xp[29];
        #pragma unroll
        for (int c = 0; c < 4; ++c) {
            float conv = p0 * wc[c][0] + p1 * wc[c][1] + p2 * wc[c][2] + p3 * wc[c][3];
            float scv  = conv * Ac[c] + Bc[c];
            int k = c * 196 + s;
            ftile[r][k] = (_Float16)((float)ftile[r][k] + scv);
        }
    }

    __syncthreads();

    // ---- WMMA GEMM, branch-free wave split: wave w takes kk = w + 4q, q=0..6 ---
    // blocks 25..27 are zero-padded; contributing nothing but keeping trips uniform
    v8f acc = {};
    const int lm = lane & 15;   // M for A-frag, N for B-frag / C/D
    const int lh = lane >> 4;   // half-select
    #pragma unroll
    for (int q = 0; q < QITER; ++q) {
        const int kbase = (wave + q * NWAVES) * 32;
        // A 16-bit layout: VGPR0-3 hold K = 8*lh + 0..7 ; VGPR4-7 hold +16
        const v8h* pa = (const v8h*)&ftile[lm][kbase + 8 * lh];
        v16h a = __builtin_shufflevector(pa[0], pa[2],
                 0,1,2,3,4,5,6,7,8,9,10,11,12,13,14,15);
        // B layout (matches A): K = 8*lh + 0..7 ; then +16
        const v8h* pb = (const v8h*)&btile[lm][kbase + 8 * lh];
        v16h b = __builtin_shufflevector(pb[0], pb[2],
                 0,1,2,3,4,5,6,7,8,9,10,11,12,13,14,15);
        acc = __builtin_amdgcn_wmma_f32_16x16x32_f16(
            false, a, false, b, (short)0, acc, false, false);
    }

    // D layout: lane holds (M = v + 8*half, N = lane%16) in VGPR v
    #pragma unroll
    for (int v = 0; v < 8; ++v)
        pltile[wave][v + lh * 8][lm] = acc[v];

    __syncthreads();

    // ---- reduce partials, bias, scale, log_softmax; one row per thread ----
    if (tid < ROWS) {
        float ls = log_scale[0];
        float vals[10];
        float mx = -1e30f;
        #pragma unroll
        for (int n = 0; n < 10; ++n) {
            float v = pltile[0][tid][n] + pltile[1][tid][n]
                    + pltile[2][tid][n] + pltile[3][tid][n];
            v = (v + lin_b[n]) * ls;
            vals[n] = v;
            mx = fmaxf(mx, v);
        }
        float se = 0.0f;
        #pragma unroll
        for (int n = 0; n < 10; ++n) se += __expf(vals[n] - mx);
        float lse = __logf(se);
        float* op = out + (size_t)(row0 + tid) * 10;
        #pragma unroll
        for (int n = 0; n < 10; ++n) op[n] = vals[n] - mx - lse;
    }
}

extern "C" void kernel_launch(void* const* d_in, const int* in_sizes, int n_in,
                              void* d_out, int out_size, void* d_ws, size_t ws_size,
                              hipStream_t stream) {
    (void)n_in; (void)out_size; (void)d_ws; (void)ws_size;
    const float* x        = (const float*)d_in[0];
    const float* conv_w   = (const float*)d_in[1];
    const float* conv_b   = (const float*)d_in[2];
    const float* bn_gamma = (const float*)d_in[3];
    const float* bn_beta  = (const float*)d_in[4];
    const float* bn_mean  = (const float*)d_in[5];
    const float* bn_var   = (const float*)d_in[6];
    const float* sw       = (const float*)d_in[7];
    const float* theta    = (const float*)d_in[8];
    const float* lin_w    = (const float*)d_in[9];
    const float* lin_b    = (const float*)d_in[10];
    const float* log_sc   = (const float*)d_in[11];
    float* out            = (float*)d_out;

    int Btotal = in_sizes[0] / 784;
    int grid   = (Btotal + ROWS - 1) / ROWS;

    quanv_fused_kernel<<<grid, NTHREADS, 0, stream>>>(
        x, conv_w, conv_b, bn_gamma, bn_beta, bn_mean, bn_var,
        sw, theta, lin_w, lin_b, log_sc, out, Btotal);
}
